// CapsuleNetwork_5574867550672
// MI455X (gfx1250) — matrix operand order repaired and checked by reading, not compiled, not run on hardware
//
#include <hip/hip_runtime.h>
#include <hip/hip_bf16.h>

// Problem sizes (fixed by the reference)
#define BB   1024
#define SS   200
#define II   4
#define HH   128
#define JJ   (II*HH)   // 512

typedef __attribute__((ext_vector_type(16))) __bf16 v16bf;
typedef __attribute__((ext_vector_type(8)))  float  v8f;

// Load two 8-float chunks at p[0..7] and p[off2..off2+7], convert to bf16.
__device__ __forceinline__ v16bf load_cvt(const float* __restrict__ p, int off2) {
    float4 x0 = *(const float4*)(p);
    float4 x1 = *(const float4*)(p + 4);
    float4 y0 = *(const float4*)(p + off2);
    float4 y1 = *(const float4*)(p + off2 + 4);
    v16bf r;
    r[0]=(__bf16)x0.x; r[1]=(__bf16)x0.y; r[2]=(__bf16)x0.z; r[3]=(__bf16)x0.w;
    r[4]=(__bf16)x1.x; r[5]=(__bf16)x1.y; r[6]=(__bf16)x1.z; r[7]=(__bf16)x1.w;
    r[8]=(__bf16)y0.x; r[9]=(__bf16)y0.y; r[10]=(__bf16)y0.z; r[11]=(__bf16)y0.w;
    r[12]=(__bf16)y1.x; r[13]=(__bf16)y1.y; r[14]=(__bf16)y1.z; r[15]=(__bf16)y1.w;
    return r;
}

// -------------------- Kernel 1: hat = per-s GEMM, bf16 WMMA --------------------
// grid.x = SS * 64 ; block = 256 (8 waves). Wave w covers N cols [w*64, w*64+64).
// C[b, j] = sum_h A[b,h] * W[s][j,h]   (A = item_eb[:, s, :], 1024x128; W row-major in h)
// Output stored transposed as hat2[b, i, s, h] bf16 (i = j>>7, h = j&127), via an
// LDS-staged transpose so global stores are coalesced b128 (128B runs per row).
#define STRD 80   // padded LDS row stride (bf16 elems): 160B, 16B-aligned, bank-spread

__global__ __launch_bounds__(256)
void capsule_hat_wmma(const float* __restrict__ item_eb,
                      const float* __restrict__ w,
                      __hip_bfloat16* __restrict__ hatws) {
    __shared__ __align__(16) __hip_bfloat16 stage[8][16][STRD];  // 20,480 B

    const int lane = threadIdx.x & 31;
    const int wave = threadIdx.x >> 5;
    const int s    = blockIdx.x >> 6;          // [0, 200)
    const int m0   = (blockIdx.x & 63) << 4;   // [0, 1024) step 16
    const int n0   = wave << 6;                // [0, 512) step 64

    const int l15 = lane & 15;
    const int hi  = lane >> 4;                 // lane half

    // A fragment base: row M = m0 + l15; lanes 0-15 hold K=k0..k0+7 & k0+16..23,
    // lanes 16-31 hold K=k0+8..15 & k0+24..31 (doc 16-bit A 16x32 layout).
    const float* abase = item_eb + (size_t)(m0 + l15) * (SS * HH) + (size_t)s * HH + (hi << 3);
    // B fragment: lane = N col; lanes 0-15 hold K=k0..k0+7 & k0+16..23, lanes 16-31 K=k0+8..15 & k0+24..31.
    const float* wbase = w + (size_t)s * (JJ * HH) + (hi << 3);

    v8f acc[4] = {};
    #pragma unroll
    for (int k0 = 0; k0 < HH; k0 += 32) {
        v16bf a = load_cvt(abase + k0, 16);
        v16bf bf[4];
        #pragma unroll
        for (int nn = 0; nn < 4; ++nn) {
            int j = n0 + nn * 16 + l15;
            bf[nn] = load_cvt(wbase + (size_t)j * HH + k0, 16);
        }
        #pragma unroll
        for (int nn = 0; nn < 4; ++nn) {
            acc[nn] = __builtin_amdgcn_wmma_f32_16x16x32_bf16(
                /*neg_a=*/false, a, /*neg_b=*/false, bf[nn],
                /*c_mod=*/(short)0, acc[nn], /*reuse_a=*/false, /*reuse_b=*/false);
        }
    }

    // C/D layout -> VGPR v holds row r = hi*8 + v (M = m0 + r), col = nn*16 + l15.
    // Stage the 16x64 bf16 tile in LDS.
    #pragma unroll
    for (int nn = 0; nn < 4; ++nn) {
        int col = nn * 16 + l15;
        #pragma unroll
        for (int v = 0; v < 8; ++v) {
            stage[wave][(hi << 3) + v][col] = __float2bfloat16(acc[nn][v]);
        }
    }
    __syncthreads();

    // Coalesced write-out: each wave's 64-col run lies within one capsule i
    // (n0 is a multiple of 64). Row r in global: 128B contiguous -> 8 x b128 chunks.
    const int iq = n0 >> 7;        // capsule index
    const int h0 = n0 & 127;       // starting h within capsule
    #pragma unroll
    for (int it = 0; it < 4; ++it) {
        int c  = it * 32 + lane;   // 0..127 chunk id
        int r  = c >> 3;           // row 0..15
        int ch = c & 7;            // 16B chunk within row
        uint4 d = *(const uint4*)&stage[wave][r][ch * 8];
        size_t off = (((size_t)(m0 + r) * II + iq) * SS + s) * HH + h0 + ch * 8;
        *(uint4*)(hatws + off) = d;
    }
}

// -------------------- Kernel 2: dynamic routing, one block per (b,i) --------------------
__global__ __launch_bounds__(256)
void capsule_route(const __hip_bfloat16* __restrict__ hatws,
                   const int* __restrict__ mask,
                   float* __restrict__ out) {
    __shared__ __align__(16) __hip_bfloat16 hatl[SS * HH];  // 51,200 B
    __shared__ float cw[SS], swv[SS], maskl[SS];
    __shared__ float interest[HH];
    __shared__ float red[256];

    const int t  = threadIdx.x;
    const int bi = blockIdx.x;      // b*4 + i
    const int b  = bi >> 2;

    // Load this capsule's hat slice (contiguous 51,200 B) into LDS as 8B chunks.
    {
        const unsigned long long* src =
            (const unsigned long long*)(hatws + (size_t)bi * (SS * HH));
        unsigned long long* dst = (unsigned long long*)hatl;
        for (int idx = t; idx < (SS * HH) / 4; idx += 256) dst[idx] = src[idx];
    }
    for (int s2 = t; s2 < SS; s2 += 256) {
        maskl[s2] = mask[b * SS + s2] ? 1.0f : 0.0f;
        cw[s2]    = 0.0f;
    }
    __syncthreads();

    for (int iter = 0; iter < 3; ++iter) {
        // ---- stable softmax over s of cw (denominator includes masked slots, as in ref) ----
        float lmax = -3.4e38f;
        for (int s2 = t; s2 < SS; s2 += 256) lmax = fmaxf(lmax, cw[s2]);
        red[t] = lmax; __syncthreads();
        #pragma unroll
        for (int off = 128; off; off >>= 1) { if (t < off) red[t] = fmaxf(red[t], red[t + off]); __syncthreads(); }
        float gmax = red[0]; __syncthreads();

        float lsum = 0.0f;
        for (int s2 = t; s2 < SS; s2 += 256) { float e = __expf(cw[s2] - gmax); swv[s2] = e; lsum += e; }
        red[t] = lsum; __syncthreads();
        #pragma unroll
        for (int off = 128; off; off >>= 1) { if (t < off) red[t] += red[t + off]; __syncthreads(); }
        float inv = 1.0f / red[0]; __syncthreads();

        // mask applied AFTER normalization (reference order)
        for (int s2 = t; s2 < SS; s2 += 256) swv[s2] = swv[s2] * inv * maskl[s2];
        __syncthreads();

        // ---- interest[h] = sum_s swv[s] * hat[s][h]  (split s-range over two thread halves) ----
        const int h    = t & 127;
        const int sBeg = (t >> 7) * 100;
        float acc = 0.0f;
        for (int s2 = sBeg; s2 < sBeg + 100; ++s2)
            acc += swv[s2] * __bfloat162float(hatl[s2 * HH + h]);
        red[t] = acc; __syncthreads();
        float vI = (t < 128) ? (red[t] + red[t + 128]) : 0.0f;
        __syncthreads();

        // ---- squash: n = sum_h vI^2 ; factor = n/(1+n)/sqrt(n+eps) ----
        red[t] = vI * vI; __syncthreads();
        #pragma unroll
        for (int off = 128; off; off >>= 1) { if (t < off) red[t] += red[t + off]; __syncthreads(); }
        float n = red[0]; __syncthreads();
        float factor = n / (1.0f + n) * rsqrtf(n + 1e-9f);
        if (t < 128) interest[t] = vI * factor;
        __syncthreads();

        if (iter < 2) {
            // ---- cw[s] += dot(hat[s,:], interest)  (per-lane h-skew to dodge bank conflicts) ----
            for (int s2 = t; s2 < SS; s2 += 256) {
                float d = 0.0f;
                const int hs = (t << 2) & 127;
                for (int k = 0; k < HH; ++k) {
                    int hh = (hs + k) & 127;
                    d += __bfloat162float(hatl[s2 * HH + hh]) * interest[hh];
                }
                cw[s2] += d;
            }
            __syncthreads();
        } else {
            if (t < 128) out[(size_t)bi * HH + t] = interest[t];
        }
    }
}

extern "C" void kernel_launch(void* const* d_in, const int* in_sizes, int n_in,
                              void* d_out, int out_size, void* d_ws, size_t ws_size,
                              hipStream_t stream) {
    const float* item_eb = (const float*)d_in[0];   // (B, S, H) f32
    const int*   mask    = (const int*)d_in[1];     // (B, S) i32
    const float* w       = (const float*)d_in[2];   // (1, S, I*H, H) f32
    float* out = (float*)d_out;                     // (B, I, H) f32
    __hip_bfloat16* hatws = (__hip_bfloat16*)d_ws;  // hat2[b,i,s,h] bf16: ~210 MB

    capsule_hat_wmma<<<dim3(SS * 64), dim3(256), 0, stream>>>(item_eb, w, hatws);
    capsule_route<<<dim3(BB * II), dim3(256), 0, stream>>>(hatws, mask, out);
}
